// PossionNet_Parametric_Cheb_71064528880235
// MI455X (gfx1250) — matrix operand order repaired and checked by reading, not compiled, not run on hardware
//
#include <hip/hip_runtime.h>
#include <hip/hip_bf16.h>
#include <stdint.h>

typedef __attribute__((ext_vector_type(16))) _Float16 v16h;
typedef __attribute__((ext_vector_type(8)))  float    v8f;
typedef __attribute__((ext_vector_type(4)))  unsigned int v4u;
typedef __attribute__((ext_vector_type(8)))  int      v8i;
typedef __attribute__((ext_vector_type(4)))  int      v4i;

#define N_NODES 50000
#define N_EDGES 800000
#define KCHEB   10

// ---------------- vectorized elementwise kernels (all sizes % 4 == 0) ----------------

__global__ void fill4_f32(float4* __restrict__ p, float v, int n4) {
    int i = blockIdx.x * blockDim.x + threadIdx.x;
    if (i < n4) p[i] = make_float4(v, v, v, v);
}

__global__ void copy4_f32(float4* __restrict__ d, const float4* __restrict__ s, int n4) {
    int i = blockIdx.x * blockDim.x + threadIdx.x;
    if (i < n4) d[i] = s[i];
}

__global__ void negcopy4_f32(float4* __restrict__ d, const float4* __restrict__ s, int n4) {
    int i = blockIdx.x * blockDim.x + threadIdx.x;
    if (i < n4) {
        float4 v = s[i];
        d[i] = make_float4(-v.x, -v.y, -v.z, -v.w);
    }
}

// ---------------- graph preprocessing ----------------

__global__ void prep_edges(const long long* __restrict__ ei, int* __restrict__ r32,
                           int* __restrict__ c32, int E) {
    int e = blockIdx.x * blockDim.x + threadIdx.x;
    if (e < E) {
        r32[e] = (int)ei[e];
        c32[e] = (int)ei[(size_t)E + e];
    }
}

__global__ void deg_kernel(const int* __restrict__ r32, float* __restrict__ deg, int E) {
    int e = blockIdx.x * blockDim.x + threadIdx.x;
    if (e < E)
        __hip_atomic_fetch_add(&deg[r32[e]], 1.0f, __ATOMIC_RELAXED, __HIP_MEMORY_SCOPE_AGENT);
}

__global__ void dis_kernel(float* __restrict__ deg, int n) {
    int i = blockIdx.x * blockDim.x + threadIdx.x;
    if (i < n) {
        float d = deg[i];
        deg[i] = (d > 0.0f) ? rsqrtf(fmaxf(d, 1e-12f)) : 0.0f;
    }
}

__global__ void anorm_kernel(const int* __restrict__ r32, const int* __restrict__ c32,
                             const float* __restrict__ dis, float* __restrict__ anorm, int E) {
    int e = blockIdx.x * blockDim.x + threadIdx.x;
    if (e < E) anorm[e] = dis[r32[e]] * dis[c32[e]];
}

// ---------------- edge scatter: dst[row] += scale * anorm[e] * src[col] ----------------
// Channel counts are powers of two -> shift instead of divide.

__global__ void scatter_vec4(float* __restrict__ dst, const float* __restrict__ src,
                             const int* __restrict__ r32, const int* __restrict__ c32,
                             const float* __restrict__ anorm, float scale,
                             int E, int C, int sh) {
    int i = blockIdx.x * blockDim.x + threadIdx.x;
    int e = i >> sh;
    if (e >= E) return;
    int c = (i - (e << sh)) << 2;
    int r  = r32[e];
    int cl = c32[e];
    float a = anorm[e] * scale;
    const float4 v = *(const float4*)(src + (size_t)cl * C + c);
    float* d = dst + (size_t)r * C + c;
    __hip_atomic_fetch_add(d + 0, a * v.x, __ATOMIC_RELAXED, __HIP_MEMORY_SCOPE_AGENT);
    __hip_atomic_fetch_add(d + 1, a * v.y, __ATOMIC_RELAXED, __HIP_MEMORY_SCOPE_AGENT);
    __hip_atomic_fetch_add(d + 2, a * v.z, __ATOMIC_RELAXED, __HIP_MEMORY_SCOPE_AGENT);
    __hip_atomic_fetch_add(d + 3, a * v.w, __ATOMIC_RELAXED, __HIP_MEMORY_SCOPE_AGENT);
}

__global__ void scatter_small(float* __restrict__ dst, const float* __restrict__ src,
                              const int* __restrict__ r32, const int* __restrict__ c32,
                              const float* __restrict__ anorm, float scale,
                              int E, int C, int sh) {
    int i = blockIdx.x * blockDim.x + threadIdx.x;
    int e = i >> sh;
    int c = i & ((1 << sh) - 1);
    if (e >= E) return;
    int r  = r32[e];
    int cl = c32[e];
    float a = anorm[e] * scale;
    __hip_atomic_fetch_add(dst + (size_t)r * C + c, a * src[(size_t)cl * C + c],
                           __ATOMIC_RELAXED, __HIP_MEMORY_SCOPE_AGENT);
}

// =====================================================================
// FAST WMMA GEMM (cin % 32 == 0, cout % 64 == 0), CIN/MT compile-time.
//   Block: 128 thr = 4 waves; block tile = 16*MT (M) x 64 (N).
//   A strip (16*MT x CIN f32) staged ONCE via Tensor Data Mover with
//   4-dword row padding (bank spread, 16B alignment); B tile
//   (CIN x 64, k-major transposed f16) staged once cooperatively.
//   Inner loop: one B fragment feeds MT back-to-back v_wmma (distinct
//   accumulators -> fills the WMMA->WMMA hazard window), no barriers.
// =====================================================================

template <int CIN, int MT>
__launch_bounds__(128)
__global__ void cheb_gemm_fast(float* __restrict__ acc, const float* __restrict__ Tx,
                               const float* __restrict__ W, int cout) {
    constexpr int ASTR = CIN + 4;   // floats per padded A row
    constexpr int BSTR = CIN + 8;   // halves per padded B row
    constexpr int MR   = 16 * MT;   // rows per block
    __shared__ __align__(16) float    Af[MR * ASTR];
    __shared__ __align__(16) _Float16 Bs[64 * BSTR];

    const int t    = threadIdx.x;
    const int wave = t >> 5;
    const int lane = t & 31;
    const int m0   = blockIdx.x * MR;
    const int nblk = blockIdx.y * 64;
    const int n0w  = nblk + wave * 16;

    // --- C fragments (one per M-subtile)
    union CF { v8f v; float f[8]; };
    CF c[MT];
    const int ccol = n0w + (lane & 15);
    const int mofs = (lane & 16) ? 8 : 0;
    #pragma unroll
    for (int mt = 0; mt < MT; ++mt)
        #pragma unroll
        for (int r = 0; r < 8; ++r)
            c[mt].f[r] = acc[(size_t)(m0 + mt * 16 + r + mofs) * cout + ccol];

    // --- stage A strip: MR rows x CIN f32 -> LDS (TDM, padded rows)
#if __has_builtin(__builtin_amdgcn_tensor_load_to_lds)
    if (t < 32) {
        const uint32_t lds = (uint32_t)(uintptr_t)(&Af[0]);
        const uint64_t ga  = (uint64_t)(uintptr_t)(Tx + (size_t)m0 * CIN);
        constexpr int pad_iv = (CIN == 32) ? 4 : (CIN == 64) ? 5 : (CIN == 128) ? 6 : 7;
        v4u g0 = { 1u, lds, (uint32_t)ga,
                   ((uint32_t)(ga >> 32) & 0x1FFFFFFu) | (2u << 30) };
        v8i g1 = { (int)((2u << 16) | (1u << 20) | ((uint32_t)pad_iv << 22) | (3u << 25)),
                   (int)(((uint32_t)CIN & 0xFFFFu) << 16),     // tensor_dim0[15:0]
                   (int)((uint32_t)MR << 16),                  // tensor_dim1 = MR
                   (int)((uint32_t)CIN << 16),                 // tile_dim0 = CIN
                   MR,                                         // tile_dim1 = MR
                   CIN,                                        // tensor_dim0_stride
                   0, 0 };
        v4i gz = { 0, 0, 0, 0 };
#if __clang_major__ >= 23
        v8i gz8 = { 0, 0, 0, 0, 0, 0, 0, 0 };
        __builtin_amdgcn_tensor_load_to_lds(g0, g1, gz, gz, gz8, 0);
#else
        __builtin_amdgcn_tensor_load_to_lds(g0, g1, gz, gz, 0);
#endif
#if __has_builtin(__builtin_amdgcn_s_wait_tensorcnt)
        __builtin_amdgcn_s_wait_tensorcnt(0);
#endif
    }
#else
    for (int idx = t; idx < MR * (CIN / 4); idx += 128) {
        int row = idx / (CIN / 4);
        int kq  = (idx % (CIN / 4)) << 2;
        *(float4*)&Af[row * ASTR + kq] =
            *(const float4*)&Tx[(size_t)(m0 + row) * CIN + kq];
    }
#endif

    // --- stage B tile transposed: Bs[n][k] f16
    for (int idx = t; idx < (CIN / 2) * 64; idx += 128) {
        int kp = idx >> 6;
        int n  = idx & 63;
        int k  = kp << 1;
        float b0 = W[(size_t)k * cout + nblk + n];
        float b1 = W[(size_t)(k + 1) * cout + nblk + n];
        union { _Float16 h[2]; uint32_t u; } p;
        p.h[0] = (_Float16)b0; p.h[1] = (_Float16)b1;
        *(uint32_t*)&Bs[n * BSTR + k] = p.u;
    }
    __syncthreads();

    const int abase = (lane & 16) ? 8 : 0;
    const int arow  = lane & 15;
    const int klane = (lane & 16) ? 8 : 0;
    const int brow  = (wave << 4) + (lane & 15);

    #pragma unroll
    for (int kk = 0; kk < CIN; kk += 32) {
        // B fragment: halves 0..7 = K kk+klane..+7, halves 8..15 = K kk+klane+16..+23 at column brow
        union { v16h v; uint4 q[2]; } b;
        b.q[0] = *(const uint4*)&Bs[brow * BSTR + kk + klane];
        b.q[1] = *(const uint4*)&Bs[brow * BSTR + kk + klane + 16];

        #pragma unroll
        for (int mt = 0; mt < MT; ++mt) {
            const int aoff = (mt * 16 + arow) * ASTR + kk + abase;
            float4 a0 = *(const float4*)&Af[aoff];
            float4 a1 = *(const float4*)&Af[aoff + 4];
            float4 a2 = *(const float4*)&Af[aoff + 16];
            float4 a3 = *(const float4*)&Af[aoff + 20];
            union { v16h v; _Float16 h[16]; } a;
            a.h[0]  = (_Float16)a0.x; a.h[1]  = (_Float16)a0.y;
            a.h[2]  = (_Float16)a0.z; a.h[3]  = (_Float16)a0.w;
            a.h[4]  = (_Float16)a1.x; a.h[5]  = (_Float16)a1.y;
            a.h[6]  = (_Float16)a1.z; a.h[7]  = (_Float16)a1.w;
            a.h[8]  = (_Float16)a2.x; a.h[9]  = (_Float16)a2.y;
            a.h[10] = (_Float16)a2.z; a.h[11] = (_Float16)a2.w;
            a.h[12] = (_Float16)a3.x; a.h[13] = (_Float16)a3.y;
            a.h[14] = (_Float16)a3.z; a.h[15] = (_Float16)a3.w;
            c[mt].v = __builtin_amdgcn_wmma_f32_16x16x32_f16(false, a.v, false, b.v,
                                                             (short)0, c[mt].v, false, false);
        }
    }

    #pragma unroll
    for (int mt = 0; mt < MT; ++mt)
        #pragma unroll
        for (int r = 0; r < 8; ++r)
            acc[(size_t)(m0 + mt * 16 + r + mofs) * cout + ccol] = c[mt].f[r];
}

// =====================================================================
// RAGGED WMMA GEMM: any cin/cout. Branch-free clamped loads + select.
// =====================================================================

#define LDSA_STRIDE 40
#define LDSB_STRIDE 40

__launch_bounds__(128)
__global__ void cheb_gemm_ragged(float* __restrict__ acc, const float* __restrict__ Tx,
                                 const float* __restrict__ W, int cin, int cout) {
    __shared__ __align__(16) _Float16 As[16 * LDSA_STRIDE];
    __shared__ __align__(16) _Float16 Bs[64 * LDSB_STRIDE];

    const int t    = threadIdx.x;
    const int wave = t >> 5;
    const int lane = t & 31;
    const int m0   = blockIdx.x * 16;
    const int nblk = blockIdx.y * 64;
    const int n0w  = nblk + wave * 16;

    union { v8f v; float f[8]; } c;
    const int ccol = n0w + (lane & 15);
    const int ccolc = min(ccol, cout - 1);
    const int mofs = (lane & 16) ? 8 : 0;
    #pragma unroll
    for (int r = 0; r < 8; ++r) {
        float v = acc[(size_t)(m0 + r + mofs) * cout + ccolc];
        c.f[r] = (ccol < cout) ? v : 0.0f;
    }

    for (int kk = 0; kk < cin; kk += 32) {
        {   // stage A 16x32: clamped loads, select-to-zero (no branches)
            int row = t >> 3;
            int k   = (t & 7) << 2;
            const float* rowp = Tx + (size_t)(m0 + row) * cin;
            union { _Float16 h[4]; uint2 u; } pk;
            #pragma unroll
            for (int j = 0; j < 4; ++j) {
                int kj = kk + k + j;
                float v = rowp[min(kj, cin - 1)];
                pk.h[j] = (_Float16)((kj < cin) ? v : 0.0f);
            }
            *(uint2*)&As[row * LDSA_STRIDE + k] = pk.u;
        }
        for (int idx = t; idx < 16 * 64; idx += 128) {  // stage B transposed, clamped
            int kp = idx >> 6;
            int n  = idx & 63;
            int k  = kp << 1;
            int nc = min(nblk + n, cout - 1);
            int k0 = min(kk + k, cin - 1);
            int k1 = min(kk + k + 1, cin - 1);
            float b0 = W[(size_t)k0 * cout + nc];
            float b1 = W[(size_t)k1 * cout + nc];
            if (nblk + n >= cout) { b0 = 0.f; b1 = 0.f; }
            if (kk + k     >= cin) b0 = 0.f;
            if (kk + k + 1 >= cin) b1 = 0.f;
            union { _Float16 h[2]; uint32_t u; } p;
            p.h[0] = (_Float16)b0; p.h[1] = (_Float16)b1;
            *(uint32_t*)&Bs[n * LDSB_STRIDE + k] = p.u;
        }
        __syncthreads();

        union { v16h v; uint4 q[2]; } a;
        {
            int abase = (lane & 16) ? 8 : 0;
            int arow  = lane & 15;
            a.q[0] = *(const uint4*)&As[arow * LDSA_STRIDE + abase];
            a.q[1] = *(const uint4*)&As[arow * LDSA_STRIDE + abase + 16];
        }
        union { v16h v; uint4 q[2]; } b;
        {
            int klane = (lane & 16) ? 8 : 0;
            int brow  = (wave << 4) + (lane & 15);
            b.q[0] = *(const uint4*)&Bs[brow * LDSB_STRIDE + klane];
            b.q[1] = *(const uint4*)&Bs[brow * LDSB_STRIDE + klane + 16];
        }
        c.v = __builtin_amdgcn_wmma_f32_16x16x32_f16(false, a.v, false, b.v,
                                                     (short)0, c.v, false, false);
        __syncthreads();
    }

    #pragma unroll
    for (int r = 0; r < 8; ++r)
        if (ccol < cout) acc[(size_t)(m0 + r + mofs) * cout + ccol] = c.f[r];
}

// ---------------- bias + optional ReLU (couts are powers of two) ----------------

__global__ void bias_act_vec4(float4* __restrict__ dst, const float4* __restrict__ acc,
                              const float* __restrict__ bias, int n4, int cmask, int relu) {
    int i = blockIdx.x * blockDim.x + threadIdx.x;
    if (i >= n4) return;
    int cb = (i << 2) & cmask;
    const float4 b = *(const float4*)&bias[cb];
    float4 v = acc[i];
    v.x += b.x; v.y += b.y; v.z += b.z; v.w += b.w;
    if (relu) {
        v.x = fmaxf(v.x, 0.f); v.y = fmaxf(v.y, 0.f);
        v.z = fmaxf(v.z, 0.f); v.w = fmaxf(v.w, 0.f);
    }
    dst[i] = v;
}

__global__ void bias_act_c1(float* __restrict__ dst, const float* __restrict__ acc,
                            const float* __restrict__ bias, int n, int relu) {
    int i = blockIdx.x * blockDim.x + threadIdx.x;
    if (i >= n) return;
    float v = acc[i] + bias[0];
    dst[i] = (relu && v < 0.f) ? 0.f : v;
}

// ---------------- host orchestration ----------------

static inline unsigned gblk(long long n, int b) { return (unsigned)((n + b - 1) / b); }

static void launch_gemm(float* acc, const float* tx, const float* w,
                        int cin, int cout, hipStream_t s) {
    if ((cin % 32) == 0 && (cout % 64) == 0) {
        switch (cin) {
            case 32:  cheb_gemm_fast<32, 5> <<<dim3(N_NODES / 80, cout / 64), 128, 0, s>>>(acc, tx, w, cout); return;
            case 64:  cheb_gemm_fast<64, 5> <<<dim3(N_NODES / 80, cout / 64), 128, 0, s>>>(acc, tx, w, cout); return;
            case 128: cheb_gemm_fast<128, 5><<<dim3(N_NODES / 80, cout / 64), 128, 0, s>>>(acc, tx, w, cout); return;
            case 256: cheb_gemm_fast<256, 1><<<dim3(N_NODES / 16, cout / 64), 128, 0, s>>>(acc, tx, w, cout); return;
            default: break;
        }
    }
    cheb_gemm_ragged<<<dim3(N_NODES / 16, (cout + 63) / 64), 128, 0, s>>>(acc, tx, w, cin, cout);
}

static void launch_scatter(float* dst, const float* src, const int* r32, const int* c32,
                           const float* anorm, float scale, int cin, hipStream_t s) {
    if ((cin & 3) == 0) {
        int sh = __builtin_ctz(cin >> 2);
        long long tot = (long long)N_EDGES << sh;
        scatter_vec4<<<gblk(tot, 256), 256, 0, s>>>(dst, src, r32, c32, anorm, scale,
                                                    N_EDGES, cin, sh);
    } else {
        int sh = (cin == 2) ? 1 : 0;
        long long tot = (long long)N_EDGES << sh;
        scatter_small<<<gblk(tot, 256), 256, 0, s>>>(dst, src, r32, c32, anorm, scale,
                                                     N_EDGES, cin, sh);
    }
}

extern "C" void kernel_launch(void* const* d_in, const int* in_sizes, int n_in,
                              void* d_out, int out_size, void* d_ws, size_t ws_size,
                              hipStream_t stream) {
    const float*     x  = (const float*)d_in[0];
    const long long* ei = (const long long*)d_in[1];
    const float* Wt[11];
    const float* Bi[11];
    for (int i = 0; i < 11; ++i) {
        Wt[i] = (const float*)d_in[2 + i];
        Bi[i] = (const float*)d_in[13 + i];
    }

    static const int CIN[11]  = {2, 32, 64, 128, 256, 128, 64, 32, 1, 32, 32};
    static const int COUT[11] = {32, 64, 128, 256, 128, 64, 32, 1, 32, 32, 1};
    static const int RELU_[11]= {1, 1, 1, 1, 1, 1, 1, 0, 1, 1, 0};

    const size_t bufElems = (size_t)N_NODES * 256;
    char* ws = (char*)d_ws;
    float* B[4];
    for (int i = 0; i < 4; ++i) B[i] = (float*)(ws + (size_t)i * bufElems * sizeof(float));
    float* dis   = (float*)(ws + 4 * bufElems * sizeof(float));
    float* anorm = dis + N_NODES;
    int*   r32   = (int*)(anorm + N_EDGES);
    int*   c32   = r32 + N_EDGES;

    // ---- edge prep + symmetric normalization
    prep_edges<<<gblk(N_EDGES, 256), 256, 0, stream>>>(ei, r32, c32, N_EDGES);
    fill4_f32 <<<gblk(N_NODES / 4, 256), 256, 0, stream>>>((float4*)dis, 0.0f, N_NODES / 4);
    deg_kernel<<<gblk(N_EDGES, 256), 256, 0, stream>>>(r32, dis, N_EDGES);
    dis_kernel<<<gblk(N_NODES, 256), 256, 0, stream>>>(dis, N_NODES);
    anorm_kernel<<<gblk(N_EDGES, 256), 256, 0, stream>>>(r32, c32, dis, anorm, N_EDGES);

    // ---- seed h = x (layer 0 input, 2 channels; 100000 floats % 4 == 0)
    copy4_f32<<<gblk(N_NODES * 2 / 4, 256), 256, 0, stream>>>((float4*)B[0], (const float4*)x,
                                                              N_NODES * 2 / 4);
    float* h = B[0];

    for (int L = 0; L < 11; ++L) {
        const int cin = CIN[L], cout = COUT[L];
        const float* W = Wt[L];
        const size_t wstride = (size_t)cin * cout;

        float* fr[3]; int nf = 0;
        for (int i = 0; i < 4; ++i) if (B[i] != h) fr[nf++] = B[i];
        float* tx0 = h;
        float* tx1 = fr[0];
        float* tx2 = fr[1];
        float* acc = fr[2];

        const long long nco = (long long)N_NODES * cout;
        const long long nci = (long long)N_NODES * cin;

        // k = 0: acc = Tx0 @ W0
        fill4_f32<<<gblk(nco / 4, 256), 256, 0, stream>>>((float4*)acc, 0.0f, (int)(nco / 4));
        launch_gemm(acc, tx0, W, cin, cout, stream);

        // k = 1: Tx1 = -A_norm @ Tx0 ; acc += Tx1 @ W1
        fill4_f32<<<gblk(nci / 4, 256), 256, 0, stream>>>((float4*)tx1, 0.0f, (int)(nci / 4));
        launch_scatter(tx1, tx0, r32, c32, anorm, -1.0f, cin, stream);
        launch_gemm(acc, tx1, W + wstride, cin, cout, stream);

        // k = 2..9: Tx2 = -2 A_norm Tx1 - Tx0 ; acc += Tx2 @ Wk ; rotate
        for (int k = 2; k < KCHEB; ++k) {
            negcopy4_f32<<<gblk(nci / 4, 256), 256, 0, stream>>>((float4*)tx2, (const float4*)tx0,
                                                                 (int)(nci / 4));
            launch_scatter(tx2, tx1, r32, c32, anorm, -2.0f, cin, stream);
            launch_gemm(acc, tx2, W + (size_t)k * wstride, cin, cout, stream);
            float* tmp = tx0; tx0 = tx1; tx1 = tx2; tx2 = tmp;
        }

        float* dst = (L == 10) ? (float*)d_out : acc;
        if (cout >= 4) {
            bias_act_vec4<<<gblk(nco / 4, 256), 256, 0, stream>>>((float4*)dst, (const float4*)acc,
                                                                  Bi[L], (int)(nco / 4),
                                                                  cout - 1, RELU_[L]);
        } else {
            bias_act_c1<<<gblk(nco, 256), 256, 0, stream>>>(dst, acc, Bi[L], (int)nco, RELU_[L]);
        }
        h = acc;
    }
}
